// MyGPT2Attention_32392643347175
// MI455X (gfx1250) — matrix operand order repaired and checked by reading, not compiled, not run on hardware
//
#include <hip/hip_runtime.h>
#include <hip/hip_bf16.h>

typedef __bf16 bf16;
typedef __attribute__((ext_vector_type(16))) __bf16        v16bf;
typedef __attribute__((ext_vector_type(8)))  float         v8f;
typedef __attribute__((ext_vector_type(4)))  unsigned int  v4u;

union AFrag { v16bf v; v4u q[2]; };

#define WMMA_BF16(a, b, c) \
  __builtin_amdgcn_wmma_f32_16x16x32_bf16(false, (a), false, (b), (short)0, (c), false, false)

constexpr int Bc = 4, Sc = 2048, Dc = 1024, Hc = 16, DH = 64;
constexpr int N3D = 3 * Dc;       // 3072
constexpr int ROWS = Bc * Sc;     // 8192

// ---------------------------------------------------------------------------
// Kernel 1a: f32 -> bf16 elementwise convert (hidden_states)
// ---------------------------------------------------------------------------
__global__ void cvt_bf16(const float* __restrict__ in, bf16* __restrict__ out, int n) {
  int i = blockIdx.x * blockDim.x + threadIdx.x;
  if (i < n) out[i] = (bf16)in[i];
}

// ---------------------------------------------------------------------------
// Kernel 1b: W [K=1024][N=3072] f32 -> W^T [N=3072][K=1024] bf16
// LDS-tiled 32x32 transpose: coalesced loads AND stores.
// ---------------------------------------------------------------------------
__global__ __launch_bounds__(256) void transpose_w(const float* __restrict__ w,
                                                   bf16* __restrict__ wt) {
  __shared__ bf16 tile[32][33];
  const int tx = threadIdx.x & 31;
  const int ty = threadIdx.x >> 5;          // 0..7
  const int n0 = blockIdx.x * 32;           // 96 tiles over N
  const int k0 = blockIdx.y * 32;           // 32 tiles over K
#pragma unroll
  for (int i = 0; i < 4; ++i)
    tile[ty + 8 * i][tx] = (bf16)w[(long)(k0 + ty + 8 * i) * N3D + n0 + tx];
  __syncthreads();
#pragma unroll
  for (int i = 0; i < 4; ++i)
    wt[(long)(n0 + ty + 8 * i) * Dc + k0 + tx] = tile[tx][ty + 8 * i];
}

// ---------------------------------------------------------------------------
// Kernel 2: QKV GEMM  C[8192,3072] = A[8192,1024] @ W[1024,3072] + bias
// Block: 256 thr (8 waves). Block tile 128x128, wave tile 32x64 (2x4 accums).
// Epilogue (part uniform per block): q -> bf16 [B,H,S,Dh]
//                                    k -> f32 present + bf16 [B,H,S,Dh]
//                                    v -> f32 present + bf16 V^T [B,H,Dh,S]
// ---------------------------------------------------------------------------
__global__ __launch_bounds__(256) void qkv_wmma(
    const bf16* __restrict__ A, const bf16* __restrict__ WT,
    const float* __restrict__ bias,
    bf16* __restrict__ qout, float* __restrict__ kout, float* __restrict__ vout,
    bf16* __restrict__ kbf, bf16* __restrict__ vtbf)
{
  const int lane = threadIdx.x & 31;
  const int wave = threadIdx.x >> 5;
  const int half = lane >> 4;
  const int l16  = lane & 15;
  const int m0 = blockIdx.y * 128 + (wave >> 1) * 32;
  const int n0 = blockIdx.x * 128 + (wave & 1) * 64;

  const v8f vzero = {0.f,0.f,0.f,0.f,0.f,0.f,0.f,0.f};
  v8f acc[2][4];
#pragma unroll
  for (int i = 0; i < 2; ++i)
#pragma unroll
    for (int j = 0; j < 4; ++j) acc[i][j] = vzero;

  for (int k0 = 0; k0 < Dc; k0 += 32) {
    // A fragments: 16x32 bf16 (ISA A layout: lane half selects K sub-block)
    AFrag af[2];
#pragma unroll
    for (int i = 0; i < 2; ++i) {
      const bf16* p = A + (long)(m0 + 16 * i + l16) * Dc + k0 + half * 8;
      af[i].q[0] = *(const v4u*)p;          // K = k0 + half*8 + 0..7
      af[i].q[1] = *(const v4u*)(p + 16);   // K = k0 + 16 + half*8 + 0..7
      if (k0 + 32 < Dc) __builtin_prefetch(p + 32, 0, 3);
    }
    // B fragments from W^T (contiguous in K)
    AFrag bfg[4];
#pragma unroll
    for (int j = 0; j < 4; ++j) {
      const bf16* p = WT + (long)(n0 + 16 * j + l16) * Dc + k0 + half * 8;
      bfg[j].q[0] = *(const v4u*)p;         // K = k0 + half*8 + 0..7
      bfg[j].q[1] = *(const v4u*)(p + 16);  // K = k0 + 16 + half*8 + 0..7
      if (k0 + 32 < Dc) __builtin_prefetch(p + 32, 0, 3);
    }
#pragma unroll
    for (int i = 0; i < 2; ++i)
#pragma unroll
      for (int j = 0; j < 4; ++j)
        acc[i][j] = WMMA_BF16(af[i].v, bfg[j].v, acc[i][j]);
  }

  // part is uniform across the whole block (128-col tile inside a 1024-col part)
  const int part = blockIdx.x >> 3;   // 0=q 1=k 2=v

#pragma unroll
  for (int j = 0; j < 4; ++j) {
    const int n = n0 + 16 * j + l16;
    const float bv = bias[n];
    const int d  = n & 1023;
    const int h  = d >> 6;
    const int dh = d & 63;
#pragma unroll
    for (int i = 0; i < 2; ++i) {
#pragma unroll
      for (int r = 0; r < 8; ++r) {
        const int row = m0 + 16 * i + half * 8 + r;   // C layout: M=(L>>4)*8+r
        const int b = row >> 11;
        const int s = row & 2047;
        const long hidx = ((long)(b * Hc + h) * Sc + s) * DH + dh;
        const float v = acc[i][j][r] + bv;
        if (part == 0) {
          qout[hidx] = (bf16)v;
        } else if (part == 1) {
          kout[hidx] = v;
          kbf[hidx]  = (bf16)v;
        } else {
          vout[hidx] = v;
          vtbf[((long)(b * Hc + h) * DH + dh) * Sc + s] = (bf16)v;  // V^T
        }
      }
    }
  }
}

// ---------------------------------------------------------------------------
// Kernel 3: causal flash attention, all-bf16 operands.
// Block = 128 thr (4 waves), each wave owns 16 query rows. Grid: (S/64, B*H).
// K read from bf16 [B,H,S,Dh]; V read from bf16 V^T [B,H,Dh,S] so every
// fragment is two contiguous global_load_b128. P staged through per-wave LDS.
// ---------------------------------------------------------------------------
__global__ __launch_bounds__(128) void attn_wmma(
    const bf16* __restrict__ Q, const bf16* __restrict__ Kb,
    const bf16* __restrict__ Vt, const float* __restrict__ mask,
    float* __restrict__ O)
{
  __shared__ __align__(16) bf16 Plds[4][16][32];   // per-wave 16x32 bf16 P tile

  const int lane = threadIdx.x & 31;
  const int wave = threadIdx.x >> 5;
  const int half = lane >> 4;
  const int l16  = lane & 15;
  const int bh = blockIdx.y;
  const int b = bh >> 4, h = bh & 15;
  const int q0 = blockIdx.x * 64 + wave * 16;

  // Q A-fragments: rows q0..q0+15, Dh=64 -> two K-steps of 32
  AFrag qa[2];
  {
    const bf16* qb = Q + ((long)bh * Sc + q0 + l16) * DH;
#pragma unroll
    for (int t = 0; t < 2; ++t) {
      qa[t].q[0] = *(const v4u*)(qb + t * 32 + half * 8);
      qa[t].q[1] = *(const v4u*)(qb + t * 32 + 16 + half * 8);
    }
  }

  const v8f vzero = {0.f,0.f,0.f,0.f,0.f,0.f,0.f,0.f};
  v8f o[4];
#pragma unroll
  for (int j = 0; j < 4; ++j) o[j] = vzero;
  float mrun[8], lrun[8];
#pragma unroll
  for (int r = 0; r < 8; ++r) { mrun[r] = -3.0e38f; lrun[r] = 0.f; }

  const bf16* kb_base = Kb + (long)bh * Sc * DH;
  const bf16* vt_base = Vt + (long)bh * DH * Sc;
  const float* mk = mask + b * Sc;

  const int nkb = (q0 + 15) / 32 + 1;   // causal limit, 32-key blocks
  for (int kb = 0; kb < nkb; ++kb) {
    const int key0 = kb * 32;

    // S = Q K^T (16 q-rows x 32 keys): two 16-key column groups
    v8f sA[2] = {vzero, vzero};
#pragma unroll
    for (int t = 0; t < 2; ++t) {        // K-step over Dh
#pragma unroll
      for (int g = 0; g < 2; ++g) {      // key group
        AFrag kf;
        const bf16* kp = kb_base + (long)(key0 + g * 16 + l16) * DH + t * 32 + half * 8;
        kf.q[0] = *(const v4u*)kp;       // dh = t*32 + half*8 + 0..7
        kf.q[1] = *(const v4u*)(kp + 16); // dh = t*32 + 16 + half*8 + 0..7
        sA[g] = WMMA_BF16(qa[t].v, kf.v, sA[g]);
      }
    }

    // scale, causal, attention-mask bias, online softmax (half-wave reduce)
    const int key_g0 = key0 + l16;
    const int key_g1 = key0 + 16 + l16;
    const float mb0 = (1.0f - mk[key_g0]) * -10000.0f;
    const float mb1 = (1.0f - mk[key_g1]) * -10000.0f;
#pragma unroll
    for (int r = 0; r < 8; ++r) {
      const int row = q0 + half * 8 + r;
      float a0 = sA[0][r] * 0.125f; if (key_g0 > row) a0 = -10000.0f; a0 += mb0;
      float a1 = sA[1][r] * 0.125f; if (key_g1 > row) a1 = -10000.0f; a1 += mb1;
      float loc = fmaxf(a0, a1);
      loc = fmaxf(loc, __shfl_xor(loc, 1, 32));
      loc = fmaxf(loc, __shfl_xor(loc, 2, 32));
      loc = fmaxf(loc, __shfl_xor(loc, 4, 32));
      loc = fmaxf(loc, __shfl_xor(loc, 8, 32));
      const float mnew  = fmaxf(mrun[r], loc);
      const float alpha = __expf(mrun[r] - mnew);
      const float p0 = __expf(a0 - mnew);
      const float p1 = __expf(a1 - mnew);
      float ls = p0 + p1;
      ls += __shfl_xor(ls, 1, 32);
      ls += __shfl_xor(ls, 2, 32);
      ls += __shfl_xor(ls, 4, 32);
      ls += __shfl_xor(ls, 8, 32);
      lrun[r] = lrun[r] * alpha + ls;
      mrun[r] = mnew;
#pragma unroll
      for (int j = 0; j < 4; ++j) o[j][r] *= alpha;
      const int rl = half * 8 + r;                 // C-layout row -> LDS row
      Plds[wave][rl][l16]      = (bf16)p0;
      Plds[wave][rl][16 + l16] = (bf16)p1;
    }
    asm volatile("s_wait_dscnt 0" ::: "memory");   // in-wave cross-lane LDS RAW

    // reload P in A-fragment layout (two ds_load_b128 per lane)
    AFrag pf;
    pf.q[0] = *(const v4u*)&Plds[wave][l16][half * 8];
    pf.q[1] = *(const v4u*)&Plds[wave][l16][16 + half * 8];

    // O += P @ V via V^T: fragments contiguous in the key dimension
#pragma unroll
    for (int j = 0; j < 4; ++j) {
      AFrag vfr;
      const bf16* vp = vt_base + (long)(j * 16 + l16) * Sc + key0 + half * 8;
      vfr.q[0] = *(const v4u*)vp;        // key = key0 + half*8 + 0..7
      vfr.q[1] = *(const v4u*)(vp + 16); // key = key0 + 16 + half*8 + 0..7
      o[j] = WMMA_BF16(pf.v, vfr.v, o[j]);
    }
  }

  // normalize and store merged heads: out[b][row][h*64 + dh]
#pragma unroll
  for (int r = 0; r < 8; ++r) {
    const int row = q0 + half * 8 + r;
    const float inv = 1.0f / lrun[r];
    float* op = O + ((long)(b * Sc + row)) * Dc + h * DH;
#pragma unroll
    for (int j = 0; j < 4; ++j) op[j * 16 + l16] = o[j][r] * inv;
  }
}

// ---------------------------------------------------------------------------
extern "C" void kernel_launch(void* const* d_in, const int* in_sizes, int n_in,
                              void* d_out, int out_size, void* d_ws, size_t ws_size,
                              hipStream_t stream) {
  (void)in_sizes; (void)n_in; (void)out_size; (void)ws_size;
  const float* hs   = (const float*)d_in[0];
  const float* mask = (const float*)d_in[1];
  const float* w    = (const float*)d_in[2];
  const float* bias = (const float*)d_in[3];

  float* out    = (float*)d_out;
  float* outAtt = out;                                   // [B,S,D]   f32
  float* outK   = out + (size_t)ROWS * Dc;               // present[0]=k [B,H,S,Dh]
  float* outV   = outK + (size_t)ROWS * Dc;              // present[1]=v

  char* ws = (char*)d_ws;
  const size_t SZ_HS = (size_t)ROWS * Dc * 2;   // 16 MB
  const size_t SZ_WT = (size_t)Dc * N3D * 2;    //  6 MB
  const size_t SZ_H  = (size_t)ROWS * Dc * 2;   // 16 MB per head-major tensor
  bf16* hs_bf = (bf16*)ws;
  bf16* wt_bf = (bf16*)(ws + SZ_HS);
  bf16* q_bf  = (bf16*)(ws + SZ_HS + SZ_WT);
  bf16* k_bf  = (bf16*)(ws + SZ_HS + SZ_WT + SZ_H);
  bf16* vt_bf = (bf16*)(ws + SZ_HS + SZ_WT + 2 * SZ_H);

  cvt_bf16<<<(ROWS * Dc) / 256, 256, 0, stream>>>(hs, hs_bf, ROWS * Dc);
  dim3 gt(N3D / 32, Dc / 32);
  transpose_w<<<gt, 256, 0, stream>>>(w, wt_bf);

  dim3 g1(N3D / 128, ROWS / 128);
  qkv_wmma<<<g1, 256, 0, stream>>>(hs_bf, wt_bf, bias, q_bf, outK, outV, k_bf, vt_bf);

  dim3 g2(Sc / 64, Bc * Hc);
  attn_wmma<<<g2, 128, 0, stream>>>(q_bf, k_bf, vt_bf, mask, outAtt);
}
